// OptionE_48455821033936
// MI455X (gfx1250) — matrix operand order repaired and checked by reading, not compiled, not run on hardware
//
#include <hip/hip_runtime.h>

typedef __bf16 bf16_t;
typedef __attribute__((ext_vector_type(8)))  __bf16 v8bf;
typedef __attribute__((ext_vector_type(16))) __bf16 v16bf;
typedef __attribute__((ext_vector_type(8)))  float  v8f;

#define HID   128
#define HEADS 4
#define HD    32
#define EDIM  16
#define EGD   32
#define QKVW  384   // fused Q|K|V row width

__device__ __forceinline__ float gelu_exact(float z) {
    return 0.5f * z * (1.0f + erff(z * 0.70710678118654752f));
}

// ---------------------------------------------------------------------------
// Weight convert + transpose:  Wt[n*K + k] = (bf16) W[k*Nc + n]
// ---------------------------------------------------------------------------
__global__ void k_convert_transpose(const float* __restrict__ W, bf16_t* __restrict__ Wt,
                                    int K, int Nc) {
    int i = blockIdx.x * blockDim.x + threadIdx.x;
    if (i >= K * Nc) return;
    int n = i / K, k = i - n * K;
    Wt[i] = (bf16_t)W[(size_t)k * Nc + n];
}

__global__ void k_f32_to_bf16(const float* __restrict__ in, bf16_t* __restrict__ out, int n) {
    int i = blockIdx.x * blockDim.x + threadIdx.x;
    if (i < n) out[i] = (bf16_t)in[i];
}

// ---------------------------------------------------------------------------
// LayerNorm (rows of 128) -> bf16.  One wave32 per row, 4 elems/lane.
// ---------------------------------------------------------------------------
__global__ void k_ln_to_bf16(const float* __restrict__ x, const float* __restrict__ g,
                             const float* __restrict__ b, bf16_t* __restrict__ out, int n) {
    int wid  = blockIdx.x * (blockDim.x >> 5) + (threadIdx.x >> 5);
    if (wid >= n) return;               // wave-uniform
    int lane = threadIdx.x & 31;
    const float4 v = *(const float4*)(x + (size_t)wid * HID + lane * 4);
    float s = v.x + v.y + v.z + v.w;
    #pragma unroll
    for (int o = 16; o; o >>= 1) s += __shfl_xor(s, o, 32);
    float mu = s * (1.0f / 128.0f);
    float dx = v.x - mu, dy = v.y - mu, dz = v.z - mu, dw = v.w - mu;
    float q = dx * dx + dy * dy + dz * dz + dw * dw;
    #pragma unroll
    for (int o = 16; o; o >>= 1) q += __shfl_xor(q, o, 32);
    float inv = rsqrtf(q * (1.0f / 128.0f) + 1e-5f);
    int j = lane * 4;
    bf16_t* po = out + (size_t)wid * HID + j;
    po[0] = (bf16_t)(dx * inv * g[j + 0] + b[j + 0]);
    po[1] = (bf16_t)(dy * inv * g[j + 1] + b[j + 1]);
    po[2] = (bf16_t)(dz * inv * g[j + 2] + b[j + 2]);
    po[3] = (bf16_t)(dw * inv * g[j + 3] + b[j + 3]);
}

// ---------------------------------------------------------------------------
// bf16 WMMA GEMM, 16x64 super-tile per wave (4 accumulators share one A frag):
//   C[M,Nc] = act(A[M,K] @ Bt[Nc,K]^T + bias)
// Nc must be a multiple of 64, M a multiple of 16, K a multiple of 32.
// act: 0=none, 1=exact GELU.  out_bf16: 0 -> f32 C, 1 -> bf16 C.
// ---------------------------------------------------------------------------
__global__ __launch_bounds__(256) void k_wmma_gemm4(
        const bf16_t* __restrict__ A, const bf16_t* __restrict__ Bt,
        const float* __restrict__ bias, void* __restrict__ Cout,
        int M, int Nc, int K, int act, int out_bf16) {
    const int wave   = blockIdx.x * (blockDim.x >> 5) + (threadIdx.x >> 5);
    const int tilesN = Nc >> 6;
    const int tileM  = wave / tilesN;
    const int tileN  = wave - tileM * tilesN;
    if (tileM * 16 >= M) return;        // wave-uniform: EXEC stays all-1s for WMMA
    const int lane    = threadIdx.x & 31;
    const int half_id = lane >> 4;      // 0: K[0..15], 1: K[16..31]
    const int rlane   = lane & 15;

    const bf16_t* pa = A  + (size_t)(tileM * 16 + rlane) * K + half_id * 16;
    const bf16_t* pb = Bt + (size_t)(tileN * 64 + rlane) * K + half_id * 16;
    const size_t  bstride = (size_t)16 * K;     // 16 columns of Bt

    v8f acc0 = {}, acc1 = {}, acc2 = {}, acc3 = {};
    for (int k0 = 0; k0 < K; k0 += 32) {
        v8bf alo = *(const v8bf*)(pa + k0);
        v8bf ahi = *(const v8bf*)(pa + k0 + 8);
        v16bf a;
        #pragma unroll
        for (int i = 0; i < 8; ++i) { a[i] = alo[i]; a[i + 8] = ahi[i]; }
        v16bf b0 = *(const v16bf*)(pb + k0);
        v16bf b1 = *(const v16bf*)(pb + bstride + k0);
        v16bf b2 = *(const v16bf*)(pb + 2 * bstride + k0);
        v16bf b3 = *(const v16bf*)(pb + 3 * bstride + k0);
        acc0 = __builtin_amdgcn_wmma_f32_16x16x32_bf16(false, a, false, b0, (short)0, acc0, false, false);
        acc1 = __builtin_amdgcn_wmma_f32_16x16x32_bf16(false, a, false, b1, (short)0, acc1, false, false);
        acc2 = __builtin_amdgcn_wmma_f32_16x16x32_bf16(false, a, false, b2, (short)0, acc2, false, false);
        acc3 = __builtin_amdgcn_wmma_f32_16x16x32_bf16(false, a, false, b3, (short)0, acc3, false, false);
    }

    #pragma unroll
    for (int t = 0; t < 4; ++t) {
        const v8f acc = (t == 0) ? acc0 : (t == 1) ? acc1 : (t == 2) ? acc2 : acc3;
        const int col = tileN * 64 + t * 16 + rlane;
        const float bv = bias ? bias[col] : 0.0f;
        #pragma unroll
        for (int i = 0; i < 8; ++i) {
            float v = acc[i] + bv;
            if (act == 1) v = gelu_exact(v);
            const size_t row = (size_t)(tileM * 16 + half_id * 8 + i);
            if (out_bf16) ((bf16_t*)Cout)[row * Nc + col] = (bf16_t)v;
            else          ((float*) Cout)[row * Nc + col] = v;
        }
    }
}

// ---------------------------------------------------------------------------
// Edge MLPs (one thread per edge):
//   ebias[e,h] = gelu(ea@Wea1+bea1) @ Wea2 + bea2
//   h1g[e,j]   = gelu(ea@Weg1+beg1)             (bf16, reused for gating)
// ---------------------------------------------------------------------------
__global__ void k_edge_mlp(const float* __restrict__ ea,
                           const float* __restrict__ Wea1, const float* __restrict__ bea1,
                           const float* __restrict__ Wea2, const float* __restrict__ bea2,
                           const float* __restrict__ Weg1, const float* __restrict__ beg1,
                           float* __restrict__ ebias, bf16_t* __restrict__ h1g, int E) {
    int e = blockIdx.x * blockDim.x + threadIdx.x;
    if (e >= E) return;
    float a[EDIM];
    #pragma unroll
    for (int i = 0; i < EDIM; ++i) a[i] = ea[(size_t)e * EDIM + i];
    float hb[EGD];
    #pragma unroll
    for (int j = 0; j < EGD; ++j) {
        float za = bea1[j], zg = beg1[j];
        #pragma unroll
        for (int i = 0; i < EDIM; ++i) {
            za += a[i] * Wea1[i * EGD + j];
            zg += a[i] * Weg1[i * EGD + j];
        }
        hb[j] = gelu_exact(za);
        h1g[(size_t)e * EGD + j] = (bf16_t)gelu_exact(zg);
    }
    #pragma unroll
    for (int h = 0; h < HEADS; ++h) {
        float z = bea2[h];
        #pragma unroll
        for (int j = 0; j < EGD; ++j) z += hb[j] * Wea2[j * HEADS + h];
        ebias[(size_t)e * HEADS + h] = z;
    }
}

// ---------------------------------------------------------------------------
// Per-edge attention logits + running segment max (one wave per edge).
// QKV fused layout: row stride 384, Q at +0, K at +128 (V at +256).
// ---------------------------------------------------------------------------
__global__ void k_edge_logits(const int* __restrict__ dstI, const int* __restrict__ srcI,
                              const float* __restrict__ QKV,
                              const float* __restrict__ ebias, float* __restrict__ logits,
                              float* __restrict__ m, int E, float scale) {
    int e = blockIdx.x * (blockDim.x >> 5) + (threadIdx.x >> 5);
    if (e >= E) return;                 // wave-uniform
    int lane = threadIdx.x & 31;
    int d = dstI[e];
    size_t qb = (size_t)d * QKVW;                 // Q[dst]
    size_t kb = (size_t)srcI[e] * QKVW + HID;     // K[src]
    float p0 = QKV[qb + 0  + lane] * QKV[kb + 0  + lane];
    float p1 = QKV[qb + 32 + lane] * QKV[kb + 32 + lane];
    float p2 = QKV[qb + 64 + lane] * QKV[kb + 64 + lane];
    float p3 = QKV[qb + 96 + lane] * QKV[kb + 96 + lane];
    #pragma unroll
    for (int o = 16; o; o >>= 1) {
        p0 += __shfl_xor(p0, o, 32);
        p1 += __shfl_xor(p1, o, 32);
        p2 += __shfl_xor(p2, o, 32);
        p3 += __shfl_xor(p3, o, 32);
    }
    if (lane < HEADS) {
        float pv = (lane == 0) ? p0 : (lane == 1) ? p1 : (lane == 2) ? p2 : p3;
        float lg = pv * scale + ebias[(size_t)e * HEADS + lane];
        logits[(size_t)e * HEADS + lane] = lg;
        atomicMax(&m[(size_t)d * HEADS + lane], lg);   // m pre-zeroed == max(.,0)
    }
}

// ---------------------------------------------------------------------------
// el = exp(logit - m[dst]); segment-sum into s.  One thread per (edge,head).
// ---------------------------------------------------------------------------
__global__ void k_edge_exp(const int* __restrict__ dstI, float* __restrict__ logits,
                           const float* __restrict__ m, float* __restrict__ s, int EH) {
    int i = blockIdx.x * blockDim.x + threadIdx.x;
    if (i >= EH) return;
    int e = i >> 2, h = i & 3;
    int d = dstI[e];
    float el = expf(logits[i] - m[(size_t)d * HEADS + h]);
    logits[i] = el;
    atomicAdd(&s[(size_t)d * HEADS + h], el);
}

// ---------------------------------------------------------------------------
// Gated, attention-weighted scatter aggregation (one wave per edge).
// Weg2 (32x128 f32) staged in LDS; atomic f32 adds land in L2 (25.6MB buffer).
// V read from fused QKV rows at +256.
// ---------------------------------------------------------------------------
__global__ __launch_bounds__(256) void k_edge_aggregate(
        const int* __restrict__ dstI, const int* __restrict__ srcI,
        const float* __restrict__ el, const float* __restrict__ s,
        const bf16_t* __restrict__ h1g, const float* __restrict__ Weg2,
        const float* __restrict__ beg2, const float* __restrict__ QKV,
        float* __restrict__ out_agg, int E) {
    __shared__ float w2[EGD * HID];
    __shared__ float b2[HID];
    for (int i = threadIdx.x; i < EGD * HID; i += blockDim.x) w2[i] = Weg2[i];
    for (int i = threadIdx.x; i < HID; i += blockDim.x) b2[i] = beg2[i];
    __syncthreads();

    int e = blockIdx.x * (blockDim.x >> 5) + (threadIdx.x >> 5);
    if (e >= E) return;                 // wave-uniform, after barrier
    int lane = threadIdx.x & 31;
    int d = dstI[e];
    size_t vb = (size_t)srcI[e] * QKVW + 2 * HID;   // V[src]

    float hv[EGD];
    #pragma unroll
    for (int k = 0; k < EGD; ++k) hv[k] = (float)h1g[(size_t)e * EGD + k];
    float attn[HEADS];
    #pragma unroll
    for (int j = 0; j < HEADS; ++j)
        attn[j] = el[(size_t)e * HEADS + j] / (s[(size_t)d * HEADS + j] + 1e-10f);

    #pragma unroll
    for (int j = 0; j < HEADS; ++j) {
        int dim = j * HD + lane;
        float gl = b2[dim];
        #pragma unroll
        for (int k = 0; k < EGD; ++k) gl += hv[k] * w2[k * HID + dim];
        float gate = 1.0f / (1.0f + expf(-gl));
        float w = attn[j] * gate * QKV[vb + dim];
        atomicAdd(&out_agg[(size_t)d * HID + dim], w);
    }
}

// ---------------------------------------------------------------------------
// FiLM + residual:  x1 = x + gamma * t + beta     (t already has bo)
// ---------------------------------------------------------------------------
__global__ void k_film_residual(const float* __restrict__ x, const float* __restrict__ t,
                                const float* __restrict__ gamma, const float* __restrict__ beta,
                                float* __restrict__ x1, int n) {
    int i = blockIdx.x * blockDim.x + threadIdx.x;
    if (i < n) x1[i] = x[i] + gamma[i] * t[i] + beta[i];
}

__global__ void k_final_add(const float* __restrict__ x1, const float* __restrict__ ffn,
                            float* __restrict__ out, int n) {
    int i = blockIdx.x * blockDim.x + threadIdx.x;
    if (i < n) out[i] = x1[i] + ffn[i];
}

// ---------------------------------------------------------------------------
extern "C" void kernel_launch(void* const* d_in, const int* in_sizes, int n_in,
                              void* d_out, int out_size, void* d_ws, size_t ws_size,
                              hipStream_t stream) {
    const float* x     = (const float*)d_in[0];
    const int*   ei    = (const int*)  d_in[1];
    const float* ea    = (const float*)d_in[2];
    const float* gamma = (const float*)d_in[3];
    const float* beta  = (const float*)d_in[4];
    const float* ln1_g = (const float*)d_in[5];
    const float* ln1_b = (const float*)d_in[6];
    const float* Wq    = (const float*)d_in[7];
    const float* Wk    = (const float*)d_in[8];
    const float* Wv    = (const float*)d_in[9];
    const float* Wo    = (const float*)d_in[10];
    const float* bo    = (const float*)d_in[11];
    const float* Wea1  = (const float*)d_in[12];
    const float* bea1  = (const float*)d_in[13];
    const float* Wea2  = (const float*)d_in[14];
    const float* bea2  = (const float*)d_in[15];
    const float* Weg1  = (const float*)d_in[16];
    const float* beg1  = (const float*)d_in[17];
    const float* Weg2  = (const float*)d_in[18];
    const float* beg2  = (const float*)d_in[19];
    const float* ln2_g = (const float*)d_in[20];
    const float* ln2_b = (const float*)d_in[21];
    const float* Wf1   = (const float*)d_in[22];
    const float* bf1   = (const float*)d_in[23];
    const float* Wf2   = (const float*)d_in[24];
    const float* bf2   = (const float*)d_in[25];

    const int N = in_sizes[0] / HID;
    const int E = in_sizes[1] / 2;
    const int* srcI = ei;
    const int* dstI = ei + E;

    char* base = (char*)d_ws;
    size_t off = 0;
    auto alloc = [&](size_t bytes) -> void* {
        void* r = base + off;
        off += (bytes + 255) & ~(size_t)255;
        return r;
    };

    bf16_t* xnb    = (bf16_t*)alloc((size_t)N * HID * 2);
    bf16_t* WqkvT  = (bf16_t*)alloc((size_t)QKVW * HID * 2);   // [384 cols][128 k]
    bf16_t* WoT    = (bf16_t*)alloc(HID * HID * 2);
    bf16_t* Wf1T   = (bf16_t*)alloc(HID * 2 * HID * 2);
    bf16_t* Wf2T   = (bf16_t*)alloc(2 * HID * HID * 2);
    float*  QKV    = (float*) alloc((size_t)N * QKVW * 4);     // fused Q|K|V rows
    float*  ebias  = (float*) alloc((size_t)E * HEADS * 4);
    bf16_t* h1g    = (bf16_t*)alloc((size_t)E * EGD * 2);
    float*  logit  = (float*) alloc((size_t)E * HEADS * 4);
    float*  mbuf   = (float*) alloc((size_t)N * HEADS * 4);
    float*  sbuf   = (float*) alloc((size_t)N * HEADS * 4);
    float*  agg    = (float*) alloc((size_t)N * HID * 4);
    bf16_t* aggb   = (bf16_t*)alloc((size_t)N * HID * 2);
    float*  tmp    = (float*) alloc((size_t)N * HID * 4);
    float*  x1     = (float*) alloc((size_t)N * HID * 4);
    bf16_t* xn2b   = (bf16_t*)alloc((size_t)N * HID * 2);
    bf16_t* hbuf   = (bf16_t*)alloc((size_t)N * 2 * HID * 2);
    float*  ffn    = (float*) alloc((size_t)N * HID * 4);
    (void)ws_size; (void)n_in; (void)out_size;

    // --- weight conversion (tiny) ---
    auto cvt = [&](const float* W, bf16_t* Wt, int K, int Nc) {
        int tot = K * Nc;
        k_convert_transpose<<<(tot + 255) / 256, 256, 0, stream>>>(W, Wt, K, Nc);
    };
    cvt(Wq, WqkvT, HID, HID);                       // cols   0..127
    cvt(Wk, WqkvT + (size_t)HID * HID, HID, HID);   // cols 128..255
    cvt(Wv, WqkvT + (size_t)2 * HID * HID, HID, HID); // cols 256..383
    cvt(Wo, WoT, HID, HID);
    cvt(Wf1, Wf1T, HID, 2 * HID);
    cvt(Wf2, Wf2T, 2 * HID, HID);

    auto gemm = [&](const bf16_t* A, const bf16_t* Bt, const float* bias, void* C,
                    int M, int Nc, int K, int act, int obf) {
        int tiles = (M / 16) * (Nc / 64);
        k_wmma_gemm4<<<(tiles + 7) / 8, 256, 0, stream>>>(A, Bt, bias, C, M, Nc, K, act, obf);
    };

    // --- LN1 + fused QKV projection (one pass over x_n) ---
    k_ln_to_bf16<<<(N + 7) / 8, 256, 0, stream>>>(x, ln1_g, ln1_b, xnb, N);
    gemm(xnb, WqkvT, nullptr, QKV, N, QKVW, HID, 0, 0);

    // --- edge MLPs ---
    k_edge_mlp<<<(E + 255) / 256, 256, 0, stream>>>(ea, Wea1, bea1, Wea2, bea2,
                                                    Weg1, beg1, ebias, h1g, E);

    // --- zero accumulators (0 bytes == 0.0f; m=0 also encodes max(.,0)) ---
    hipMemsetAsync(mbuf, 0, (size_t)N * HEADS * 4, stream);
    hipMemsetAsync(sbuf, 0, (size_t)N * HEADS * 4, stream);
    hipMemsetAsync(agg, 0, (size_t)N * HID * 4, stream);

    // --- sparse softmax attention + gated aggregation ---
    const float scale = 0.17677669529663687f;   // 32^-0.5
    k_edge_logits<<<(E + 7) / 8, 256, 0, stream>>>(dstI, srcI, QKV, ebias, logit,
                                                   mbuf, E, scale);
    k_edge_exp<<<(E * HEADS + 255) / 256, 256, 0, stream>>>(dstI, logit, mbuf, sbuf,
                                                            E * HEADS);
    k_edge_aggregate<<<(E + 7) / 8, 256, 0, stream>>>(dstI, srcI, logit, sbuf, h1g,
                                                      Weg2, beg2, QKV, agg, E);

    // --- output projection + FiLM + residual ---
    k_f32_to_bf16<<<(N * HID + 255) / 256, 256, 0, stream>>>(agg, aggb, N * HID);
    gemm(aggb, WoT, bo, tmp, N, HID, HID, 0, 0);
    k_film_residual<<<(N * HID + 255) / 256, 256, 0, stream>>>(x, tmp, gamma, beta,
                                                               x1, N * HID);

    // --- FFN ---
    k_ln_to_bf16<<<(N + 7) / 8, 256, 0, stream>>>(x1, ln2_g, ln2_b, xn2b, N);
    gemm(xn2b, Wf1T, bf1, hbuf, N, 2 * HID, HID, 1, 1);     // GELU, bf16 out
    gemm(hbuf, Wf2T, bf2, ffn, N, HID, 2 * HID, 0, 0);
    k_final_add<<<(N * HID + 255) / 256, 256, 0, stream>>>(x1, ffn, (float*)d_out,
                                                           N * HID);
}
